// SelfBalancingExpertsV3_4252017623357
// MI455X (gfx1250) — compile-verified, matrix operand order checked by reading, not run on hardware
//
#include <hip/hip_runtime.h>
#include <hip/hip_bf16.h>

// ---------------------------------------------------------------------------
// SelfBalancingExpertsV3 — MI455X (gfx1250, wave32, WMMA) implementation.
//  * top-K gathered expert MLP (4x fewer FLOPs than dense reference)
//  * bf16 WMMA (v_wmma_f32_16x16x32_bf16), fp32 accumulate
//  * weights pre-converted to transposed bf16 so GEMM staging is pure
//    K-contiguous b128 traffic; async global->LDS double buffering
//  (runtime GEMM dims on purpose: constexpr trip counts made clang fully
//   unroll the K loop and spill all accumulators to scratch)
// ---------------------------------------------------------------------------

typedef __attribute__((ext_vector_type(16))) __bf16 v16bf;
typedef __attribute__((ext_vector_type(8)))  __bf16 v8bf;
typedef __attribute__((ext_vector_type(4)))  __bf16 v4bf;
typedef __attribute__((ext_vector_type(8)))  float  v8f;
typedef __attribute__((ext_vector_type(4)))  int    v4i;

#define T_TOK 4096
#define DDIM  1024
#define EEXP  8
#define HDIM  2048

#define BM 128
#define BN 128
#define BK 32
#define LDT 40   // LDS row stride in bf16 elements (80 B, keeps 16B alignment)

// ---- CDNA5 async global->LDS path (guarded; falls back to sync copies) ----
#if defined(__has_builtin)
#if __has_builtin(__builtin_amdgcn_global_load_async_to_lds_b128) && \
    __has_builtin(__builtin_amdgcn_s_wait_asynccnt)
#define USE_ASYNC_LDS 1
#endif
#endif
#ifndef USE_ASYNC_LDS
#define USE_ASYNC_LDS 0
#endif

#if USE_ASYNC_LDS
typedef __attribute__((address_space(1))) v4i* glb_v4i_p;   // global (device)
typedef __attribute__((address_space(3))) v4i* lds_v4i_p;   // LDS (shared)
__device__ __forceinline__ void async_b128(const void* g, void* l) {
  __builtin_amdgcn_global_load_async_to_lds_b128((glb_v4i_p)g, (lds_v4i_p)l, 0, 0);
}
#endif

// --------------------------- gating (exact fp32) ---------------------------
__global__ void gate_kernel(const float* __restrict__ x,
                            const float* __restrict__ gW,
                            const float* __restrict__ gb,
                            const float* __restrict__ gtemp,
                            float* __restrict__ scores,
                            int*   __restrict__ topk,
                            float* __restrict__ usage,
                            int*   __restrict__ counts) {
  int wave = threadIdx.x >> 5;
  int lane = threadIdx.x & 31;
  int t = blockIdx.x * 8 + wave;
  if (t >= T_TOK) return;
  const float* xr = x + (size_t)t * DDIM;
  float acc[8];
#pragma unroll
  for (int e = 0; e < 8; ++e) acc[e] = 0.f;
  for (int it = 0; it < DDIM / 32; ++it) {
    int d = lane + it * 32;
    float xv = xr[d];
    const float4 w0 = *(const float4*)(gW + (size_t)d * 8);
    const float4 w1 = *(const float4*)(gW + (size_t)d * 8 + 4);
    acc[0] += xv * w0.x; acc[1] += xv * w0.y; acc[2] += xv * w0.z; acc[3] += xv * w0.w;
    acc[4] += xv * w1.x; acc[5] += xv * w1.y; acc[6] += xv * w1.z; acc[7] += xv * w1.w;
  }
#pragma unroll
  for (int off = 16; off >= 1; off >>= 1)
#pragma unroll
    for (int e = 0; e < 8; ++e) acc[e] += __shfl_down(acc[e], off);
  if (lane == 0) {
    float invT = 1.f / gtemp[0];
    float lg[8], mx = -3.4e38f;
#pragma unroll
    for (int e = 0; e < 8; ++e) { lg[e] = (acc[e] + gb[e]) * invT; mx = fmaxf(mx, lg[e]); }
    float s = 0.f;
#pragma unroll
    for (int e = 0; e < 8; ++e) { lg[e] = expf(lg[e] - mx); s += lg[e]; }
    float inv = 1.f / s;
#pragma unroll
    for (int e = 0; e < 8; ++e) { lg[e] *= inv; scores[(size_t)t * 8 + e] = lg[e]; }
    int i0 = 0; float v0 = lg[0];
#pragma unroll
    for (int e = 1; e < 8; ++e) if (lg[e] > v0) { v0 = lg[e]; i0 = e; }
    int i1 = -1; float v1 = -3.4e38f;
#pragma unroll
    for (int e = 0; e < 8; ++e) if (e != i0 && lg[e] > v1) { v1 = lg[e]; i1 = e; }
    topk[t * 2]     = i0;
    topk[t * 2 + 1] = i1;
    atomicAdd(usage + i0, 1.0f);
    atomicAdd(counts + i0, 1);
    atomicAdd(counts + i1, 1);
  }
}

// ------------------------ EM iterations + KL loss --------------------------
__global__ void em_loss_kernel(const float* __restrict__ scores,
                               const float* __restrict__ usage,
                               float* __restrict__ pOut,
                               float* __restrict__ lossOut) {
  __shared__ float sp[8];
  __shared__ float scnt[8];
  int tid = threadIdx.x;
  if (tid < 8) { sp[tid] = 1.0f / 8.0f; scnt[tid] = 0.f; }
  __syncthreads();
  for (int iter = 0; iter < 5; ++iter) {
    float lp[8], part[8];
#pragma unroll
    for (int e = 0; e < 8; ++e) { lp[e] = sp[e]; part[e] = 0.f; }
    for (int t = tid; t < T_TOK; t += 256) {
      float ea[8]; float s = 0.f;
#pragma unroll
      for (int e = 0; e < 8; ++e) { ea[e] = scores[(size_t)t * 8 + e] * lp[e]; s += ea[e]; }
      float inv = 1.f / (s + 1e-8f);
#pragma unroll
      for (int e = 0; e < 8; ++e) part[e] += ea[e] * inv;
    }
#pragma unroll
    for (int e = 0; e < 8; ++e) atomicAdd(&scnt[e], part[e]);
    __syncthreads();
    if (tid == 0) {
      float tot = 0.f;
      for (int e = 0; e < 8; ++e) tot += scnt[e];
      float inv = 1.f / (tot + 1e-8f);
      for (int e = 0; e < 8; ++e) sp[e] = scnt[e] * inv;
    }
    __syncthreads();
    if (tid < 8) scnt[tid] = 0.f;
    __syncthreads();
  }
  if (tid < 8) pOut[tid] = sp[tid];
  if (tid == 0) {
    float a[8]; float s = 0.f;
    for (int e = 0; e < 8; ++e) { a[e] = usage[e] * (1.0f / (float)T_TOK) + 1e-8f; s += a[e]; }
    float u = 1.0f / 8.0f, kl = 0.f;
    for (int e = 0; e < 8; ++e) kl += u * (logf(u) - logf(a[e] / s));
    lossOut[0] = 0.1f * kl;
  }
}

// -------------------- per-expert segment build (scan+scatter) --------------
__global__ void init_kernel(float* usage, int* counts, int* cursors) {
  int i = threadIdx.x;
  if (i < 8) { usage[i] = 0.f; counts[i] = 0; cursors[i] = 0; }
}

__global__ void scan_kernel(const int* __restrict__ counts, int* __restrict__ offs) {
  if (threadIdx.x == 0) {
    int s = 0;
    for (int e = 0; e < 8; ++e) { offs[e] = s; s += counts[e]; }
  }
}

__global__ void scatter_kernel(const int* __restrict__ topk,
                               const float* __restrict__ scores,
                               const float* __restrict__ p,
                               const int* __restrict__ offs,
                               int* __restrict__ cursors,
                               int* __restrict__ permT,
                               float* __restrict__ permW) {
  int t = blockIdx.x * 256 + threadIdx.x;
  if (t >= T_TOK) return;
#pragma unroll
  for (int k = 0; k < 2; ++k) {
    int e = topk[t * 2 + k];
    int pos = atomicAdd(&cursors[e], 1);
    int slot = offs[e] + pos;
    permT[slot] = t;
    permW[slot] = scores[(size_t)t * 8 + e] * p[e];
  }
}

// -------------------- fp32 -> bf16 conversion / transpose ------------------
__global__ void cvt_kernel(const float* __restrict__ in, __bf16* __restrict__ out) {
  int i = blockIdx.x * 256 + threadIdx.x;
  float4 v = ((const float4*)in)[i];
  v4bf c;
  c[0] = (__bf16)v.x; c[1] = (__bf16)v.y; c[2] = (__bf16)v.z; c[3] = (__bf16)v.w;
  *(v4bf*)(out + (size_t)i * 4) = c;
}

// out[z][c][r] = (bf16) in[z][r][c];  R,C multiples of 32
__global__ void transpose_cvt_kernel(const float* __restrict__ in,
                                     __bf16* __restrict__ outp,
                                     int R, int C) {
  __shared__ float tile[32][33];
  const float* src = in   + (size_t)blockIdx.z * R * C;
  __bf16*      dst = outp + (size_t)blockIdx.z * R * C;
  int c0 = blockIdx.x * 32, r0 = blockIdx.y * 32;
  int tx = threadIdx.x & 31, ty = threadIdx.x >> 5;   // 32 x 8
#pragma unroll
  for (int i = 0; i < 4; ++i) {
    int r = ty + 8 * i;
    tile[r][tx] = src[(size_t)(r0 + r) * C + c0 + tx];
  }
  __syncthreads();
#pragma unroll
  for (int i = 0; i < 4; ++i) {
    int c = ty + 8 * i;
    dst[(size_t)(c0 + c) * R + r0 + tx] = (__bf16)tile[tx][c];
  }
}

// -------------------------- WMMA fragment loader ---------------------------
// 16-bit A/B fragment for V_WMMA_*_16X16X32: lanes 0-15 hold row (lane), K in
// {0..7,16..23}; lanes 16-31 hold row (lane-16), K in {8..15,24..31}. Two
// contiguous 8-element (16 B) runs per lane -> two ds_load_b128.
__device__ __forceinline__ v16bf frag_load(const __bf16* __restrict__ tile, int row) {
  int lane = threadIdx.x & 31;
  int kb = (lane & 16) >> 1;            // 0 or 8
  const __bf16* base = tile + row * LDT + kb;
  v8bf lo = *(const v8bf*)(base);
  v8bf hi = *(const v8bf*)(base + 16);
  v16bf r;
#pragma unroll
  for (int i = 0; i < 8; ++i) { r[i] = lo[i]; r[8 + i] = hi[i]; }
  return r;
}

// ------------------------------- tiled GEMM --------------------------------
// All operands bf16; B is pre-transposed ([N][K] row-major) so both tiles are
// staged as K-contiguous b128 chunks (async global->LDS, double buffered).
// MODE 0: h_s  = relu(xbf @ sW1t^T + sb1)           (out bf16)
// MODE 1: out  = 0.1*(h_s @ sW2t^T + sb2)           (plain f32 store)
// MODE 2: h_g  = relu(xbf[perm] @ W1t[e]^T + b1[e]) (gathered rows, out bf16)
// MODE 3: out += w * (h_g @ W2t[e]^T + b2[e])       (f32 atomic scatter)
template <int MODE>
__global__ void moe_gemm(const __bf16* __restrict__ Abf,
                         const __bf16* __restrict__ Bt,
                         const float* __restrict__ biasSrc,
                         __bf16* __restrict__ Obf,
                         float* __restrict__ Of32,
                         const int* __restrict__ counts,
                         const int* __restrict__ offs,
                         const int* __restrict__ permT,
                         const float* __restrict__ permW,
                         int M, int N, int Ksz) {
  __shared__ __bf16 tiles[4][BM * LDT];   // [buf0:A,B][buf1:A,B]  (~40 KB)
  __shared__ int    sMap[BM];

  const int tid  = threadIdx.x;
  const int lane = tid & 31;
  const int wave = tid >> 5;
  const int wmBase = (wave & 3) * 32;     // 4 waves along M
  const int wnBase = (wave >> 2) * 64;    // 2 waves along N
  const int lm = lane & 15;
  const int hi8 = (lane & 16) ? 8 : 0;

  const int e = (MODE >= 2) ? blockIdx.z : 0;
  int n = M, segOff = 0;
  if (MODE >= 2) {
    n = counts[e];
    segOff = offs[e];
    if ((int)blockIdx.x * BM >= n) return;   // uniform early exit
  }
  const int row0 = blockIdx.x * BM;
  const int col0 = blockIdx.y * BN;
  const __bf16* Bp   = Bt      + (size_t)e * Ksz * N;  // [N][K] per expert
  const float*  bias = biasSrc + (size_t)e * N;

  if (MODE == 2) {
    if (tid < BM) {
      int r = row0 + tid;
      sMap[tid] = (r < n) ? permT[segOff + r] : 0;   // clamp to valid row
    }
    __syncthreads();
  }

  // ---- per-thread staging assignment: 2 A-chunks + 2 B-chunks of 8 bf16 ----
  const int m0 = tid >> 2,       c0e = (tid & 3) * 8;   // rows 0..63
  const int m1 = (tid >> 2) + 64;                       // rows 64..127
  const int aoff0 = m0 * LDT + c0e, aoff1 = m1 * LDT + c0e;

  auto arow = [&](int m) -> int {
    if (MODE == 0 || MODE == 1) return row0 + m;
    if (MODE == 2) return sMap[m];
    int r = row0 + m; if (r >= n) r = n - 1;
    return segOff + r;
  };
  const __bf16* ap0 = Abf + (size_t)arow(m0) * Ksz + c0e;
  const __bf16* ap1 = Abf + (size_t)arow(m1) * Ksz + c0e;
  const __bf16* bp0 = Bp + (size_t)(col0 + m0) * Ksz + c0e;
  const __bf16* bp1 = Bp + (size_t)(col0 + m1) * Ksz + c0e;

  auto stage = [&](int buf) {
    __bf16* tA = tiles[buf * 2];
    __bf16* tB = tiles[buf * 2 + 1];
#if USE_ASYNC_LDS
    async_b128(ap0, tA + aoff0);
    async_b128(ap1, tA + aoff1);
    async_b128(bp0, tB + aoff0);
    async_b128(bp1, tB + aoff1);
#else
    *(v8bf*)(tA + aoff0) = *(const v8bf*)ap0;
    *(v8bf*)(tA + aoff1) = *(const v8bf*)ap1;
    *(v8bf*)(tB + aoff0) = *(const v8bf*)bp0;
    *(v8bf*)(tB + aoff1) = *(const v8bf*)bp1;
#endif
    ap0 += BK; ap1 += BK; bp0 += BK; bp1 += BK;
  };

  v8f acc[2][4];
#pragma unroll
  for (int wm = 0; wm < 2; ++wm)
#pragma unroll
    for (int wn = 0; wn < 4; ++wn)
#pragma unroll
      for (int i = 0; i < 8; ++i) acc[wm][wn][i] = 0.f;

  auto compute = [&](int buf) {
    const __bf16* tA = tiles[buf * 2];
    const __bf16* tB = tiles[buf * 2 + 1];
    v16bf af[2], bfg[4];
#pragma unroll
    for (int wm = 0; wm < 2; ++wm) af[wm] = frag_load(tA, wmBase + wm * 16 + lm);
#pragma unroll
    for (int wn = 0; wn < 4; ++wn) bfg[wn] = frag_load(tB, wnBase + wn * 16 + lm);
#pragma unroll
    for (int wm = 0; wm < 2; ++wm)
#pragma unroll
      for (int wn = 0; wn < 4; ++wn)
        acc[wm][wn] = __builtin_amdgcn_wmma_f32_16x16x32_bf16(
            false, af[wm], false, bfg[wn], (short)0, acc[wm][wn], false, false);
  };

  const int nk = Ksz / BK;   // runtime on purpose (prevents full unroll+spill)
  stage(0);
  int cur = 0;
  // main loop: branch-free body, last iteration peeled
  for (int kt = 0; kt < nk - 1; ++kt) {
    stage(cur ^ 1);                           // prefetch next tile into other buf
#if USE_ASYNC_LDS
    __builtin_amdgcn_s_wait_asynccnt(4);      // retire current buffer's 4 loads
#endif
    __syncthreads();                          // current buffer fully staged
    compute(cur);
    cur ^= 1;
    __syncthreads();                          // all reads done before overwrite
  }
#if USE_ASYNC_LDS
  __builtin_amdgcn_s_wait_asynccnt(0);
#endif
  __syncthreads();
  compute(cur);

  // ---- epilogue ----
#pragma unroll
  for (int wn = 0; wn < 4; ++wn) {
    int col = col0 + wnBase + wn * 16 + lm;
    float bv = bias[col];
#pragma unroll
    for (int wm = 0; wm < 2; ++wm) {
#pragma unroll
      for (int i = 0; i < 8; ++i) {
        int mrow = wmBase + wm * 16 + hi8 + i;
        int r = row0 + mrow;
        float v = acc[wm][wn][i] + bv;
        if (MODE == 0) {
          Obf[(size_t)r * N + col] = (__bf16)fmaxf(v, 0.f);
        } else if (MODE == 1) {
          Of32[(size_t)r * N + col] = 0.1f * v;
        } else if (MODE == 2) {
          if (r < n) Obf[(size_t)(segOff + r) * N + col] = (__bf16)fmaxf(v, 0.f);
        } else {
          if (r < n) {
            int tkn = permT[segOff + r];
            float w = permW[segOff + r];
            atomicAdd(Of32 + (size_t)tkn * N + col, w * v);
          }
        }
      }
    }
  }
}

// ------------------------------- launcher ----------------------------------
extern "C" void kernel_launch(void* const* d_in, const int* in_sizes, int n_in,
                              void* d_out, int out_size, void* d_ws, size_t ws_size,
                              hipStream_t stream) {
  const float* x     = (const float*)d_in[0];
  const float* gW    = (const float*)d_in[1];
  const float* gb    = (const float*)d_in[2];
  const float* gtemp = (const float*)d_in[3];
  const float* W1    = (const float*)d_in[4];
  const float* b1    = (const float*)d_in[5];
  const float* W2    = (const float*)d_in[6];
  const float* b2    = (const float*)d_in[7];
  const float* sW1   = (const float*)d_in[8];
  const float* sb1   = (const float*)d_in[9];
  const float* sW2   = (const float*)d_in[10];
  const float* sb2   = (const float*)d_in[11];
  float* out = (float*)d_out;

  // workspace carve-up (~130 MB total; 256B aligned chunks)
  char* base = (char*)d_ws;
  size_t off = 0;
  auto carve = [&](size_t bytes) -> char* {
    char* p = base + off;
    off = (off + bytes + 255) & ~(size_t)255;
    return p;
  };
  float* scores  = (float*)carve((size_t)T_TOK * EEXP * 4);
  float* pbuf    = (float*)carve(8 * 4);
  float* usage   = (float*)carve(8 * 4);
  int*   counts  = (int*)  carve(8 * 4);
  int*   offsArr = (int*)  carve(8 * 4);
  int*   cursors = (int*)  carve(8 * 4);
  int*   topk    = (int*)  carve((size_t)T_TOK * 2 * 4);
  int*   permT   = (int*)  carve((size_t)T_TOK * 2 * 4);
  float* permW   = (float*)carve((size_t)T_TOK * 2 * 4);
  __bf16* h_s    = (__bf16*)carve((size_t)T_TOK * HDIM * 2);
  __bf16* h_g    = (__bf16*)carve((size_t)T_TOK * 2 * HDIM * 2);
  __bf16* xbf    = (__bf16*)carve((size_t)T_TOK * DDIM * 2);
  __bf16* W1t    = (__bf16*)carve((size_t)EEXP * DDIM * HDIM * 2);
  __bf16* W2t    = (__bf16*)carve((size_t)EEXP * DDIM * HDIM * 2);
  __bf16* sW1t   = (__bf16*)carve((size_t)DDIM * HDIM * 2);
  __bf16* sW2t   = (__bf16*)carve((size_t)DDIM * HDIM * 2);
  (void)in_sizes; (void)n_in; (void)out_size; (void)ws_size;

  // ---- precision / layout prep (memory bound, ~11us at HBM peak) ----
  cvt_kernel<<<(T_TOK * DDIM / 4) / 256, 256, 0, stream>>>(x, xbf);
  transpose_cvt_kernel<<<dim3(HDIM / 32, DDIM / 32, EEXP), 256, 0, stream>>>(W1, W1t, DDIM, HDIM);
  transpose_cvt_kernel<<<dim3(DDIM / 32, HDIM / 32, EEXP), 256, 0, stream>>>(W2, W2t, HDIM, DDIM);
  transpose_cvt_kernel<<<dim3(HDIM / 32, DDIM / 32, 1),    256, 0, stream>>>(sW1, sW1t, DDIM, HDIM);
  transpose_cvt_kernel<<<dim3(DDIM / 32, HDIM / 32, 1),    256, 0, stream>>>(sW2, sW2t, HDIM, DDIM);

  // ---- routing (exact fp32) ----
  init_kernel<<<1, 32, 0, stream>>>(usage, counts, cursors);
  gate_kernel<<<T_TOK / 8, 256, 0, stream>>>(x, gW, gb, gtemp, scores, topk, usage, counts);
  em_loss_kernel<<<1, 256, 0, stream>>>(scores, usage, pbuf, out + (size_t)T_TOK * DDIM);
  scan_kernel<<<1, 32, 0, stream>>>(counts, offsArr);
  scatter_kernel<<<T_TOK / 256, 256, 0, stream>>>(topk, scores, pbuf, offsArr, cursors, permT, permW);

  // ---- shared MLP: plain-stores every d_out element BEFORE expert scatter --
  moe_gemm<0><<<dim3(T_TOK / BM, HDIM / BN), 256, 0, stream>>>(
      xbf, sW1t, sb1, h_s, nullptr, nullptr, nullptr, nullptr, nullptr,
      T_TOK, HDIM, DDIM);
  moe_gemm<1><<<dim3(T_TOK / BM, DDIM / BN), 256, 0, stream>>>(
      h_s, sW2t, sb2, nullptr, out, nullptr, nullptr, nullptr, nullptr,
      T_TOK, DDIM, HDIM);

  // ---- expert MLP over gathered (token,expert) segments ----
  moe_gemm<2><<<dim3(64, HDIM / BN, EEXP), 256, 0, stream>>>(
      xbf, W1t, b1, h_g, nullptr, counts, offsArr, permT, permW,
      T_TOK * 2, HDIM, DDIM);
  moe_gemm<3><<<dim3(64, DDIM / BN, EEXP), 256, 0, stream>>>(
      h_g, W2t, b2, nullptr, out, counts, offsArr, permT, permW,
      T_TOK * 2, DDIM, HDIM);
}
